// DiTBlockWithMoE_5239860101490
// MI455X (gfx1250) — matrix operand order repaired and checked by reading, not compiled, not run on hardware
//
#include <hip/hip_runtime.h>
#include <hip/hip_bf16.h>
#include <math.h>

#define B_    2
#define S_    2048
#define CTX_  512
#define DIM_  1536
#define NH_   12
#define FFN_  6144
#define NE_   4
#define TOPK_ 2
#define HD_   128
#define EPS_  1e-6f

typedef __attribute__((ext_vector_type(16))) __bf16 v16bf;
typedef __attribute__((ext_vector_type(8)))  __bf16 v8bf;
typedef __attribute__((ext_vector_type(8)))  float  v8f;

static __device__ __forceinline__ v16bf mkfrag(v8bf a, v8bf b) {
  return __builtin_shufflevector(a, b, 0,1,2,3,4,5,6,7,8,9,10,11,12,13,14,15);
}
static __device__ __forceinline__ v8f wmma_bf16(v16bf a, v16bf b, v8f c) {
  // D = A(16x32) * B(32x16) + C, fp32 accumulate
  return __builtin_amdgcn_wmma_f32_16x16x32_bf16(false, a, false, b, (short)0, c, false, false);
}
static __device__ __forceinline__ float gelu_tanh(float x) {
  float x3 = x * x * x;
  return 0.5f * x * (1.f + tanhf(0.7978845608028654f * (x + 0.044715f * x3)));
}

// ---------------------------------------------------------------- convert
__global__ void k_f32_to_bf16(const float* __restrict__ s, __bf16* __restrict__ d, int n) {
  int i = blockIdx.x * blockDim.x + threadIdx.x;
  if (i < n) d[i] = (__bf16)s[i];
}

// ------------------------------------------------- LayerNorm (+modulation)
// dst = LN(src) [* gamma + beta] [* (1+scale_mod) + shift_mod]   (bf16 out)
__global__ __launch_bounds__(256) void k_ln_mod(
    const float* __restrict__ src, const float* __restrict__ tmod,
    const float* __restrict__ modw, const float* __restrict__ gamma,
    const float* __restrict__ beta, int shiftIdx, int scaleIdx,
    __bf16* __restrict__ dst)
{
  __shared__ float r1[256];
  __shared__ float r2[256];
  int t = blockIdx.x;
  int b = t / S_;
  const float* x = src + (size_t)t * DIM_;
  float a = 0.f, q = 0.f;
#pragma unroll
  for (int i = 0; i < DIM_ / 256; i++) {
    float v = x[threadIdx.x + i * 256];
    a += v; q += v * v;
  }
  r1[threadIdx.x] = a; r2[threadIdx.x] = q;
  __syncthreads();
  for (int st = 128; st > 0; st >>= 1) {
    if (threadIdx.x < st) { r1[threadIdx.x] += r1[threadIdx.x + st]; r2[threadIdx.x] += r2[threadIdx.x + st]; }
    __syncthreads();
  }
  float mean = r1[0] * (1.f / DIM_);
  float var  = r2[0] * (1.f / DIM_) - mean * mean;
  float rs = rsqrtf(var + EPS_);
  __bf16* y = dst + (size_t)t * DIM_;
#pragma unroll
  for (int i = 0; i < DIM_ / 256; i++) {
    int d = threadIdx.x + i * 256;
    float v = (x[d] - mean) * rs;
    if (gamma) v = v * gamma[d] + beta[d];
    if (scaleIdx >= 0) {
      float sc = modw[(size_t)scaleIdx * DIM_ + d] + tmod[((size_t)b * 6 + scaleIdx) * DIM_ + d];
      float sh = modw[(size_t)shiftIdx * DIM_ + d] + tmod[((size_t)b * 6 + shiftIdx) * DIM_ + d];
      v = v * (1.f + sc) + sh;
    }
    y[d] = (__bf16)v;
  }
}

// ------------------------------------------------------- RMSNorm (+RoPE)
__global__ __launch_bounds__(256) void k_rms_rope(
    const float* __restrict__ src, const float* __restrict__ w,
    const float* __restrict__ cosb, const float* __restrict__ sinb,
    int rope, __bf16* __restrict__ dst)
{
  __shared__ float r2[256];
  int t = blockIdx.x;
  const float* x = src + (size_t)t * DIM_;
  float q = 0.f;
#pragma unroll
  for (int i = 0; i < DIM_ / 256; i++) { float v = x[threadIdx.x + i * 256]; q += v * v; }
  r2[threadIdx.x] = q;
  __syncthreads();
  for (int st = 128; st > 0; st >>= 1) {
    if (threadIdx.x < st) r2[threadIdx.x] += r2[threadIdx.x + st];
    __syncthreads();
  }
  float rs = rsqrtf(r2[0] * (1.f / DIM_) + EPS_);
  __bf16* y = dst + (size_t)t * DIM_;
  if (rope) {
    int s = t % S_;
#pragma unroll
    for (int i = 0; i < DIM_ / 512; i++) {           // 3 rotation pairs / thread
      int p = threadIdx.x + i * 256;                  // 0..767 ; head = p/64, fi = p%64
      int fi = p & 63;
      float c  = cosb[(size_t)s * (HD_ / 2) + fi];
      float sn = sinb[(size_t)s * (HD_ / 2) + fi];
      float va = x[2 * p]     * rs * w[2 * p];
      float vb = x[2 * p + 1] * rs * w[2 * p + 1];
      y[2 * p]     = (__bf16)(va * c - vb * sn);
      y[2 * p + 1] = (__bf16)(va * sn + vb * c);
    }
  } else {
#pragma unroll
    for (int i = 0; i < DIM_ / 256; i++) {
      int d = threadIdx.x + i * 256;
      y[d] = (__bf16)(x[d] * rs * w[d]);
    }
  }
}

// --------------------------------------------------------- WMMA GEMM
// C[M,N] = A[M,K](bf16) @ W[N,K]^T(bf16) + bias ; optional GELU
// 128x128 block tile, 256 threads (8 waves: 4 M x 2 N), each wave 32x64
// (2x4 WMMA tiles, 8 wmma per 32-K step). Global->LDS software pipelined.
// Output: fp32 row-major (OUTF), bf16 row-major (OUTB), or bf16 transposed
// head-major [B][DIM][SV] (TRANSB, for attention V).
template<bool GELU, bool OUTF, bool OUTB, bool TRANSB>
__global__ __launch_bounds__(256) void k_gemm(
    const __bf16* __restrict__ A, const __bf16* __restrict__ W,
    const float* __restrict__ bias, float* __restrict__ Cf,
    __bf16* __restrict__ Cb, int M, int N, int K, int SV)
{
  __shared__ __bf16 As[128][40];   // pitch 40 (80B) to spread banks
  __shared__ __bf16 Ws[128][40];
  int tid = threadIdx.x, lane = tid & 31, wvid = tid >> 5;
  int wm = wvid >> 1, wn = wvid & 1;       // 4 waves along M, 2 along N
  int bm = blockIdx.y * 128, bn = blockIdx.x * 128;
  int lm = lane & 15, hi = lane >> 4;
  v8f zero = {0.f,0.f,0.f,0.f,0.f,0.f,0.f,0.f};
  v8f acc[2][4];
#pragma unroll
  for (int i = 0; i < 2; i++)
#pragma unroll
    for (int j = 0; j < 4; j++) acc[i][j] = zero;

  // staging-register mapping: chunk c = tid + i*256 -> row c>>2, col (c&3)*8
  v8bf pa[2], pw[2];
#pragma unroll
  for (int i = 0; i < 2; i++) {
    int c = tid + i * 256;
    int row = c >> 2, col = (c & 3) * 8;
    pa[i] = *(const v8bf*)(A + (size_t)(bm + row) * K + col);
    pw[i] = *(const v8bf*)(W + (size_t)(bn + row) * K + col);
  }

  for (int kb = 0; kb < K; kb += 32) {
#pragma unroll
    for (int i = 0; i < 2; i++) {
      int c = tid + i * 256;
      int row = c >> 2, col = (c & 3) * 8;
      *(v8bf*)&As[row][col] = pa[i];
      *(v8bf*)&Ws[row][col] = pw[i];
    }
    __syncthreads();
    if (kb + 32 < K) {
#pragma unroll
      for (int i = 0; i < 2; i++) {
        int c = tid + i * 256;
        int row = c >> 2, col = (c & 3) * 8;
        pa[i] = *(const v8bf*)(A + (size_t)(bm + row) * K + kb + 32 + col);
        pw[i] = *(const v8bf*)(W + (size_t)(bn + row) * K + kb + 32 + col);
      }
    }
    v16bf af[2], wf[4];
#pragma unroll
    for (int mi = 0; mi < 2; mi++) {
      int r = wm * 32 + mi * 16 + lm;   // A frag: K = hi*8+i / 16+hi*8+i
      af[mi] = mkfrag(*(const v8bf*)&As[r][hi * 8], *(const v8bf*)&As[r][16 + hi * 8]);
    }
#pragma unroll
    for (int ni = 0; ni < 4; ni++) {
      int n = wn * 64 + ni * 16 + lm;   // B frag: K = hi*8+i / 16+hi*8+i
      wf[ni] = mkfrag(*(const v8bf*)&Ws[n][hi * 8], *(const v8bf*)&Ws[n][16 + hi * 8]);
    }
#pragma unroll
    for (int mi = 0; mi < 2; mi++)
#pragma unroll
      for (int ni = 0; ni < 4; ni++)
        acc[mi][ni] = wmma_bf16(af[mi], wf[ni], acc[mi][ni]);
    __syncthreads();
  }
#pragma unroll
  for (int mi = 0; mi < 2; mi++)
#pragma unroll
    for (int ni = 0; ni < 4; ni++)
#pragma unroll
      for (int r = 0; r < 8; r++) {
        int row = bm + wm * 32 + mi * 16 + hi * 8 + r;
        int col = bn + wn * 64 + ni * 16 + lm;
        float v = acc[mi][ni][r] + bias[col];
        if (GELU) v = gelu_tanh(v);
        if (OUTF) Cf[(size_t)row * N + col] = v;
        if (OUTB) Cb[(size_t)row * N + col] = (__bf16)v;
        if (TRANSB) {
          int bb = row / SV, srow = row - bb * SV;    // token -> (batch, seq)
          Cb[((size_t)bb * DIM_ + col) * SV + srow] = (__bf16)v;
        }
      }
}

// --------------------------------------------------------- Flash attention
// One wave per 16-query tile; QK^T + online softmax + PV, all bf16 WMMA.
// V is pre-transposed: Vt[B][DIM][Skv] (key index contiguous) so PV
// B-fragments are two contiguous global_load_b128 per 16x16 tile.
__global__ __launch_bounds__(128) void k_flash(
    const __bf16* __restrict__ Q, const __bf16* __restrict__ K,
    const __bf16* __restrict__ Vt, __bf16* __restrict__ O,
    int Sq, int Skv, float scale)
{
  __shared__ __bf16 PtAll[4][16][40];    // P staging: row=q, col=key
  int lane = threadIdx.x & 31, wvid = threadIdx.x >> 5;
  int lm = lane & 15, hi = lane >> 4;
  int qt = blockIdx.x * 4 + wvid;
  int h = blockIdx.y, b = blockIdx.z;
  int q0 = qt * 16;
  const __bf16* Qp  = Q  + ((size_t)b * Sq + q0) * DIM_ + h * HD_;
  const __bf16* Kp  = K  + (size_t)b * Skv * DIM_ + h * HD_;
  const __bf16* Vtp = Vt + ((size_t)b * DIM_ + h * HD_) * Skv;
  __bf16 (*pt)[40] = PtAll[wvid];

  v16bf qf[4];
#pragma unroll
  for (int dk = 0; dk < 4; dk++) {
    const __bf16* p = Qp + (size_t)lm * DIM_ + dk * 32 + hi * 8;
    qf[dk] = mkfrag(*(const v8bf*)p, *(const v8bf*)(p + 16));
  }
  v8f zero = {0.f,0.f,0.f,0.f,0.f,0.f,0.f,0.f};
  v8f oacc[8];
#pragma unroll
  for (int t = 0; t < 8; t++) oacc[t] = zero;
  float mrow[8], lrow[8];
#pragma unroll
  for (int r = 0; r < 8; r++) { mrow[r] = -1e30f; lrow[r] = 0.f; }

  for (int kk = 0; kk < Skv; kk += 32) {
    // scores: two 16x16 tiles (keys kk..kk+15, kk+16..kk+31)
    v8f sc0 = zero, sc1 = zero;
    {
      const __bf16* kp0 = Kp + (size_t)(kk + lm) * DIM_;
      const __bf16* kp1 = Kp + (size_t)(kk + 16 + lm) * DIM_;
#pragma unroll
      for (int dk = 0; dk < 4; dk++) {
        v16bf b0 = mkfrag(*(const v8bf*)(kp0 + dk * 32 + hi * 8),
                          *(const v8bf*)(kp0 + dk * 32 + hi * 8 + 16));
        sc0 = wmma_bf16(qf[dk], b0, sc0);
        v16bf b1 = mkfrag(*(const v8bf*)(kp1 + dk * 32 + hi * 8),
                          *(const v8bf*)(kp1 + dk * 32 + hi * 8 + 16));
        sc1 = wmma_bf16(qf[dk], b1, sc1);
      }
    }
    // online softmax (row stats reduced over 16 lanes of each half-wave)
    float alpha[8];
#pragma unroll
    for (int r = 0; r < 8; r++) {
      float a0 = sc0[r] * scale, a1 = sc1[r] * scale;
      float mx = fmaxf(a0, a1);
#pragma unroll
      for (int m = 1; m < 16; m <<= 1) mx = fmaxf(mx, __shfl_xor(mx, m, 32));
      float mnew = fmaxf(mrow[r], mx);
      float al = __expf(mrow[r] - mnew);
      float p0 = __expf(a0 - mnew), p1 = __expf(a1 - mnew);
      float rsum = p0 + p1;
#pragma unroll
      for (int m = 1; m < 16; m <<= 1) rsum += __shfl_xor(rsum, m, 32);
      lrow[r] = lrow[r] * al + rsum;
      mrow[r] = mnew;
      alpha[r] = al;
      int row = hi * 8 + r;
      pt[row][lm]      = (__bf16)p0;
      pt[row][16 + lm] = (__bf16)p1;
    }
#pragma unroll
    for (int t = 0; t < 8; t++)
#pragma unroll
      for (int r = 0; r < 8; r++) oacc[t][r] = oacc[t][r] * alpha[r];
    // P as A-fragment (16x32), V^T rows as B-fragments, accumulate O
    v16bf pf = mkfrag(*(const v8bf*)&pt[lm][hi * 8], *(const v8bf*)&pt[lm][16 + hi * 8]);
#pragma unroll
    for (int t = 0; t < 8; t++) {
      const __bf16* vp = Vtp + (size_t)(t * 16 + lm) * Skv + kk + hi * 8;
      v16bf vfrag = mkfrag(*(const v8bf*)vp, *(const v8bf*)(vp + 16));
      oacc[t] = wmma_bf16(pf, vfrag, oacc[t]);
    }
  }
  __bf16* Op = O + ((size_t)b * Sq + q0) * DIM_ + h * HD_;
#pragma unroll
  for (int t = 0; t < 8; t++)
#pragma unroll
    for (int r = 0; r < 8; r++) {
      int row = hi * 8 + r;
      Op[(size_t)row * DIM_ + t * 16 + lm] = (__bf16)(oacc[t][r] / lrow[r]);
    }
}

// ------------------------------------------------- residual (+gate) add
__global__ void k_resid(float* __restrict__ out, const float* __restrict__ t,
                        const float* __restrict__ tmod, const float* __restrict__ modw,
                        int gateIdx, int n)
{
  int i = blockIdx.x * blockDim.x + threadIdx.x;
  if (i >= n) return;
  float g = 1.f;
  if (gateIdx >= 0) {
    int d = i % DIM_;
    int b = i / (S_ * DIM_);
    g = modw[(size_t)gateIdx * DIM_ + d] + tmod[((size_t)b * 6 + gateIdx) * DIM_ + d];
  }
  out[i] += g * t[i];
}

// -------------------------------------------------------- MoE combine
__global__ void k_moe_combine(float* __restrict__ acc, const float* __restrict__ t,
                              const float* __restrict__ ew, const int* __restrict__ idx,
                              int e, int n)
{
  int i = blockIdx.x * blockDim.x + threadIdx.x;
  if (i >= n) return;
  int tok = i / DIM_;
  float w = 0.f;
#pragma unroll
  for (int k = 0; k < TOPK_; k++)
    if (idx[tok * TOPK_ + k] == e) w += ew[tok * TOPK_ + k];
  acc[i] += w * t[i];
}

// ===================================================================
extern "C" void kernel_launch(void* const* d_in, const int* in_sizes, int n_in,
                              void* d_out, int out_size, void* d_ws, size_t ws_size,
                              hipStream_t stream)
{
  (void)in_sizes; (void)n_in; (void)out_size; (void)ws_size;
  const float* x       = (const float*)d_in[0];
  const float* context = (const float*)d_in[1];
  const float* t_mod   = (const float*)d_in[2];
  const float* fcos    = (const float*)d_in[3];
  const float* fsin    = (const float*)d_in[4];
  const float* ew      = (const float*)d_in[5];
  const int*   topk    = (const int*)d_in[6];
  const float* sa_q_w = (const float*)d_in[7];   const float* sa_q_b = (const float*)d_in[8];
  const float* sa_k_w = (const float*)d_in[9];   const float* sa_k_b = (const float*)d_in[10];
  const float* sa_v_w = (const float*)d_in[11];  const float* sa_v_b = (const float*)d_in[12];
  const float* sa_o_w = (const float*)d_in[13];  const float* sa_o_b = (const float*)d_in[14];
  const float* sa_nq_w = (const float*)d_in[15]; const float* sa_nk_w = (const float*)d_in[16];
  const float* ca_q_w = (const float*)d_in[17];  const float* ca_q_b = (const float*)d_in[18];
  const float* ca_k_w = (const float*)d_in[19];  const float* ca_k_b = (const float*)d_in[20];
  const float* ca_v_w = (const float*)d_in[21];  const float* ca_v_b = (const float*)d_in[22];
  const float* ca_o_w = (const float*)d_in[23];  const float* ca_o_b = (const float*)d_in[24];
  const float* ca_nq_w = (const float*)d_in[25]; const float* ca_nk_w = (const float*)d_in[26];
  const float* norm3_w = (const float*)d_in[27]; const float* norm3_b = (const float*)d_in[28];
  const float* ffn_w1 = (const float*)d_in[29];  const float* ffn_b1 = (const float*)d_in[30];
  const float* ffn_w2 = (const float*)d_in[31];  const float* ffn_b2 = (const float*)d_in[32];
  const float* modw   = (const float*)d_in[33];
  const float* moe_w  = (const float*)d_in[34];  const float* moe_b  = (const float*)d_in[35];
  float* out = (float*)d_out;

  const size_t DD  = (size_t)DIM_ * DIM_;
  const size_t FD  = (size_t)FFN_ * DIM_;
  const size_t TOK = (size_t)B_ * S_;     // 4096 tokens
  const size_t CT  = (size_t)B_ * CTX_;   // 1024 ctx tokens

  char* wsb = (char*)d_ws;
  size_t off = 0;
  auto allocB = [&](size_t elems) -> __bf16* {
    __bf16* p = (__bf16*)(wsb + off); off += ((elems * 2 + 255) & ~(size_t)255); return p;
  };
  auto allocF = [&](size_t elems) -> float* {
    float* p = (float*)(wsb + off); off += ((elems * 4 + 255) & ~(size_t)255); return p;
  };
  __bf16* wq   = allocB(DD);  __bf16* wk   = allocB(DD);
  __bf16* wvw  = allocB(DD);  __bf16* wo   = allocB(DD);
  __bf16* wcqw = allocB(DD);  __bf16* wckw = allocB(DD);
  __bf16* wcvw = allocB(DD);  __bf16* wcow = allocB(DD);
  __bf16* wf1  = allocB(FD);  __bf16* wf2  = allocB(FD);
  __bf16* wmoe = allocB((size_t)NE_ * DD);
  __bf16* ctxb = allocB(CT * DIM_);
  __bf16* hb   = allocB(TOK * DIM_);
  __bf16* qb   = allocB(TOK * DIM_);
  __bf16* kb   = allocB(TOK * DIM_);
  __bf16* vtb  = allocB(TOK * DIM_);      // self V, transposed [B][DIM][S]
  __bf16* ab   = allocB(TOK * DIM_);
  __bf16* cvtb = allocB(CT * DIM_);       // cross V, transposed [B][DIM][CTX]
  __bf16* midb = allocB(TOK * FFN_);
  float*  tmpf  = allocF(TOK * DIM_);
  float*  ffacc = allocF(TOK * DIM_);

  auto conv = [&](const float* s, __bf16* d, size_t n) {
    k_f32_to_bf16<<<dim3((unsigned)((n + 255) / 256)), 256, 0, stream>>>(s, d, (int)n);
  };
  auto gemmF = [&](const __bf16* A, const __bf16* W, const float* bias, float* Cf,
                   int M, int N, int Kd) {
    k_gemm<false, true, false, false><<<dim3(N / 128, M / 128), 256, 0, stream>>>(
        A, W, bias, Cf, nullptr, M, N, Kd, 0);
  };
  auto gemmVT = [&](const __bf16* A, const __bf16* W, const float* bias, __bf16* Cb,
                    int M, int N, int Kd, int SV) {
    k_gemm<false, false, false, true><<<dim3(N / 128, M / 128), 256, 0, stream>>>(
        A, W, bias, nullptr, Cb, M, N, Kd, SV);
  };
  auto gemmG = [&](const __bf16* A, const __bf16* W, const float* bias, __bf16* Cb,
                   int M, int N, int Kd) {
    k_gemm<true, false, true, false><<<dim3(N / 128, M / 128), 256, 0, stream>>>(
        A, W, bias, nullptr, Cb, M, N, Kd, 0);
  };
  const int NTOT = (int)(TOK * DIM_);
  const dim3 gElem((NTOT + 255) / 256);
  const float scale = 0.08838834764831845f;   // 1/sqrt(128)

  // weights -> bf16 (L2-resident: 192MB covers everything)
  conv(sa_q_w, wq, DD);   conv(sa_k_w, wk, DD);
  conv(sa_v_w, wvw, DD);  conv(sa_o_w, wo, DD);
  conv(ca_q_w, wcqw, DD); conv(ca_k_w, wckw, DD);
  conv(ca_v_w, wcvw, DD); conv(ca_o_w, wcow, DD);
  conv(ffn_w1, wf1, FD);  conv(ffn_w2, wf2, FD);
  conv(moe_w, wmoe, (size_t)NE_ * DD);
  conv(context, ctxb, CT * DIM_);

  // residual accumulator lives in d_out
  hipMemcpyAsync(out, x, TOK * DIM_ * sizeof(float), hipMemcpyDeviceToDevice, stream);

  // ---- self attention ----
  k_ln_mod<<<dim3((unsigned)TOK), 256, 0, stream>>>(out, t_mod, modw, nullptr, nullptr, 0, 1, hb);
  gemmF(hb, wq, sa_q_b, tmpf, (int)TOK, DIM_, DIM_);
  k_rms_rope<<<dim3((unsigned)TOK), 256, 0, stream>>>(tmpf, sa_nq_w, fcos, fsin, 1, qb);
  gemmF(hb, wk, sa_k_b, tmpf, (int)TOK, DIM_, DIM_);
  k_rms_rope<<<dim3((unsigned)TOK), 256, 0, stream>>>(tmpf, sa_nk_w, fcos, fsin, 1, kb);
  gemmVT(hb, wvw, sa_v_b, vtb, (int)TOK, DIM_, DIM_, S_);
  k_flash<<<dim3(S_ / 64, NH_, B_), 128, 0, stream>>>(qb, kb, vtb, ab, S_, S_, scale);
  gemmF(ab, wo, sa_o_b, tmpf, (int)TOK, DIM_, DIM_);
  k_resid<<<gElem, 256, 0, stream>>>(out, tmpf, t_mod, modw, 2, NTOT);

  // ---- cross attention ----
  k_ln_mod<<<dim3((unsigned)TOK), 256, 0, stream>>>(out, t_mod, modw, norm3_w, norm3_b, -1, -1, hb);
  gemmF(hb, wcqw, ca_q_b, tmpf, (int)TOK, DIM_, DIM_);
  k_rms_rope<<<dim3((unsigned)TOK), 256, 0, stream>>>(tmpf, ca_nq_w, fcos, fsin, 0, qb);
  gemmF(ctxb, wckw, ca_k_b, tmpf, (int)CT, DIM_, DIM_);
  k_rms_rope<<<dim3((unsigned)CT), 256, 0, stream>>>(tmpf, ca_nk_w, fcos, fsin, 0, kb);
  gemmVT(ctxb, wcvw, ca_v_b, cvtb, (int)CT, DIM_, DIM_, CTX_);
  k_flash<<<dim3(S_ / 64, NH_, B_), 128, 0, stream>>>(qb, kb, cvtb, ab, S_, CTX_, scale);
  gemmF(ab, wcow, ca_o_b, tmpf, (int)TOK, DIM_, DIM_);
  k_resid<<<gElem, 256, 0, stream>>>(out, tmpf, t_mod, modw, -1, NTOT);

  // ---- FFN + MoE ----
  k_ln_mod<<<dim3((unsigned)TOK), 256, 0, stream>>>(out, t_mod, modw, nullptr, nullptr, 3, 4, hb);
  gemmG(hb, wf1, ffn_b1, midb, (int)TOK, FFN_, DIM_);
  gemmF(midb, wf2, ffn_b2, ffacc, (int)TOK, DIM_, FFN_);
  for (int e = 0; e < NE_; e++) {
    gemmF(hb, wmoe + (size_t)e * DD, moe_b + (size_t)e * DIM_, tmpf, (int)TOK, DIM_, DIM_);
    k_moe_combine<<<gElem, 256, 0, stream>>>(ffacc, tmpf, ew, topk, e, NTOT);
  }
  k_resid<<<gElem, 256, 0, stream>>>(out, ffacc, t_mod, modw, 5, NTOT);
}
